// APPNP__58188216926735
// MI455X (gfx1250) — compile-verified, matrix operand order checked by reading, not run on hardware
//
#include <hip/hip_runtime.h>
#include <hip/hip_bf16.h>

typedef __attribute__((ext_vector_type(16))) __bf16 v16bf;
typedef __attribute__((ext_vector_type(8)))  __bf16 v8bf;
typedef __attribute__((ext_vector_type(8)))  float  v8f;

#define IN_DIM  512
#define HID_DIM 256
#define OUT_DIM 64
#define ALPHA   0.1f

static __device__ __forceinline__ v16bf join16(v8bf lo, v8bf hi) {
    return __builtin_shufflevector(lo, hi, 0,1,2,3,4,5,6,7,8,9,10,11,12,13,14,15);
}

// ---------------- weight prep: f32 [K,N] -> bf16 transposed [N,K] ----------------
__global__ void k_prep_w(const float* __restrict__ W, __bf16* __restrict__ Wt,
                         int K, int Ncol) {
    int t = blockIdx.x * blockDim.x + threadIdx.x;
    if (t >= K * Ncol) return;
    int n = t / K;
    int k = t - n * K;
    Wt[t] = (__bf16)W[(size_t)k * Ncol + n];
}

// ---------------- degree / normalization ----------------
__global__ void k_deg_init(unsigned* __restrict__ deg, int n) {
    int t = blockIdx.x * blockDim.x + threadIdx.x;
    if (t < n) deg[t] = 1u;  // self loop
}
__global__ void k_deg_count(unsigned* __restrict__ deg, const int* __restrict__ col, int E) {
    int t = blockIdx.x * blockDim.x + threadIdx.x;
    if (t < E) atomicAdd(&deg[col[t]], 1u);
}
__global__ void k_dinv(void* __restrict__ buf, int n) {
    int t = blockIdx.x * blockDim.x + threadIdx.x;
    if (t >= n) return;
    unsigned d = ((unsigned*)buf)[t];
    ((float*)buf)[t] = rsqrtf((float)d);
}

// ---------------- GEMM1: hid = relu(x @ W1 + b1), bf16 WMMA ----------------
// block = 320 threads = 10 waves; wave -> one 16-row tile, full 256 columns.
// Double-buffered LDS W1 slices (one barrier per k-step) + register
// double-buffered B fragments so WMMA never waits on a just-issued ds_load.
__global__ __launch_bounds__(320) void k_gemm1(const float* __restrict__ x,
                                               const __bf16* __restrict__ Wt1,  // [256,512]
                                               const float* __restrict__ b1,
                                               __bf16* __restrict__ hid,        // [N,256]
                                               int ntiles) {
    __shared__ __bf16 lds_w[2][256 * 40];   // two 32-wide K slices, padded stride 40
    const int tid  = threadIdx.x;
    const int wave = tid >> 5;
    const int lane = tid & 31;
    const int mtile = blockIdx.x * 10 + wave;
    const int mt = (mtile < ntiles) ? mtile : (ntiles - 1);
    const bool wr = (mtile < ntiles);
    const int m    = lane & 15;
    const int koff = (lane >> 4) << 3;      // 0 or 8
    const int row  = mt * 16 + m;

    v8f c[16] = {};

    // ---- prologue: stage slice 0 into LDS buffer 0, load A regs for k0=0 ----
    if (tid < 256) {
        const __bf16* src = Wt1 + (size_t)tid * IN_DIM;
        __bf16* dst = &lds_w[0][tid * 40];
        *(v8bf*)(dst)      = *(const v8bf*)(src);
        *(v8bf*)(dst + 8)  = *(const v8bf*)(src + 8);
        *(v8bf*)(dst + 16) = *(const v8bf*)(src + 16);
        *(v8bf*)(dst + 24) = *(const v8bf*)(src + 24);
    }
    const float* xr0 = x + (size_t)row * IN_DIM + koff;
    float4 f0 = *(const float4*)(xr0);
    float4 f1 = *(const float4*)(xr0 + 4);
    float4 f2 = *(const float4*)(xr0 + 16);
    float4 f3 = *(const float4*)(xr0 + 20);
    __syncthreads();

    for (int ks = 0; ks < IN_DIM / 32; ++ks) {
        const int cur = ks & 1;
        const int nxt = cur ^ 1;
        const bool havenext = (ks < IN_DIM / 32 - 1);
        const int k0 = ks * 32;

        // 1) kick off global loads for the NEXT W1 slice (into registers)
        v8bf g0 = {}, g1 = {}, g2 = {}, g3 = {};
        if (havenext && tid < 256) {
            const __bf16* src = Wt1 + (size_t)tid * IN_DIM + k0 + 32;
            g0 = *(const v8bf*)(src);
            g1 = *(const v8bf*)(src + 8);
            g2 = *(const v8bf*)(src + 16);
            g3 = *(const v8bf*)(src + 24);
        }
        // 2) kick off global loads for the NEXT A fragment
        float4 n0, n1, n2, n3;
        if (havenext) {
            const float* xr = x + (size_t)row * IN_DIM + k0 + 32 + koff;
            n0 = *(const float4*)(xr);
            n1 = *(const float4*)(xr + 4);
            n2 = *(const float4*)(xr + 16);
            n3 = *(const float4*)(xr + 20);
        }

        // 3) build A fragment (f32 -> bf16)
        v16bf a;
        a[0]=(__bf16)f0.x; a[1]=(__bf16)f0.y; a[2]=(__bf16)f0.z; a[3]=(__bf16)f0.w;
        a[4]=(__bf16)f1.x; a[5]=(__bf16)f1.y; a[6]=(__bf16)f1.z; a[7]=(__bf16)f1.w;
        a[8]=(__bf16)f2.x; a[9]=(__bf16)f2.y; a[10]=(__bf16)f2.z; a[11]=(__bf16)f2.w;
        a[12]=(__bf16)f3.x; a[13]=(__bf16)f3.y; a[14]=(__bf16)f3.z; a[15]=(__bf16)f3.w;

        // 4) 16 WMMAs with register double-buffered B fragments from LDS
        const __bf16* bp = &lds_w[cur][m * 40 + koff];
        v8bf blo = *(const v8bf*)(bp);
        v8bf bhi = *(const v8bf*)(bp + 16);
        #pragma unroll
        for (int nt = 0; nt < 16; ++nt) {
            v8bf plo = blo, phi = bhi;
            if (nt < 15) {
                const __bf16* wp = &lds_w[cur][((nt + 1) * 16 + m) * 40 + koff];
                blo = *(const v8bf*)(wp);
                bhi = *(const v8bf*)(wp + 16);
            }
            v16bf b = join16(plo, phi);
            c[nt] = __builtin_amdgcn_wmma_f32_16x16x32_bf16(
                        false, a, false, b, (short)0, c[nt], false, false);
        }

        // 5) commit next slice to the other LDS buffer, rotate A regs, barrier
        if (havenext) {
            if (tid < 256) {
                __bf16* dst = &lds_w[nxt][tid * 40];
                *(v8bf*)(dst)      = g0;
                *(v8bf*)(dst + 8)  = g1;
                *(v8bf*)(dst + 16) = g2;
                *(v8bf*)(dst + 24) = g3;
            }
            f0 = n0; f1 = n1; f2 = n2; f3 = n3;
            __syncthreads();
        }
    }

    if (wr) {
        #pragma unroll
        for (int nt = 0; nt < 16; ++nt) {
            const int coln = nt * 16 + m;
            const float bias = b1[coln];
            #pragma unroll
            for (int j = 0; j < 8; ++j) {
                int r = mt * 16 + j + ((lane >> 4) << 3);
                float v = c[nt][j] + bias;
                v = v > 0.f ? v : 0.f;
                hid[(size_t)r * HID_DIM + coln] = (__bf16)v;
            }
        }
    }
}

// ---------------- GEMM2: h0 = hid @ W2 + b2, bf16 WMMA ----------------
__global__ __launch_bounds__(256) void k_gemm2(const __bf16* __restrict__ hid,  // [N,256]
                                               const __bf16* __restrict__ Wt2,  // [64,256]
                                               const float* __restrict__ b2,
                                               float* __restrict__ h0,          // [N,64]
                                               int ntiles) {
    const int wave = (blockIdx.x * blockDim.x + threadIdx.x) >> 5;
    if (wave >= ntiles) return;                 // wave-uniform exit
    const int lane = threadIdx.x & 31;
    const int m    = lane & 15;
    const int koff = (lane >> 4) << 3;
    const int row  = wave * 16 + m;

    v8f c[4] = {};
    #pragma unroll
    for (int k0 = 0; k0 < HID_DIM; k0 += 32) {
        const __bf16* ap = hid + (size_t)row * HID_DIM + k0 + koff;
        v16bf a = join16(*(const v8bf*)(ap), *(const v8bf*)(ap + 16));
        #pragma unroll
        for (int nt = 0; nt < 4; ++nt) {
            const __bf16* wp = Wt2 + (size_t)(nt * 16 + m) * HID_DIM + k0 + koff;
            v16bf b = join16(*(const v8bf*)(wp), *(const v8bf*)(wp + 16));
            c[nt] = __builtin_amdgcn_wmma_f32_16x16x32_bf16(
                        false, a, false, b, (short)0, c[nt], false, false);
        }
    }
    #pragma unroll
    for (int nt = 0; nt < 4; ++nt) {
        const int coln = nt * 16 + m;
        const float bias = b2[coln];
        #pragma unroll
        for (int j = 0; j < 8; ++j) {
            int r = wave * 16 + j + ((lane >> 4) << 3);
            h0[(size_t)r * OUT_DIM + coln] = c[nt][j] + bias;
        }
    }
}

// ---------------- propagation: h = 0.9 * segsum(msg) + 0.1 * h0 ----------------
__global__ void k_prop_init(float* __restrict__ hout, const float* __restrict__ hin,
                            const float* __restrict__ h0, const float* __restrict__ dinv,
                            int n64) {
    int t = blockIdx.x * blockDim.x + threadIdx.x;
    if (t >= n64) return;
    int i = t >> 6;
    float di = dinv[i];
    hout[t] = ALPHA * h0[t] + (1.0f - ALPHA) * di * di * hin[t];
}

// 16 threads per edge, 4 features each (float4 gather, f32 atomic scatter)
__global__ void k_prop_scatter(float* __restrict__ hout, const float* __restrict__ hin,
                               const int* __restrict__ row, const int* __restrict__ col,
                               const float* __restrict__ dinv, int E) {
    int t = blockIdx.x * blockDim.x + threadIdx.x;
    int e = t >> 4;
    if (e >= E) return;
    int q = (t & 15) << 2;
    int r = row[e], c = col[e];
    float w = (1.0f - ALPHA) * dinv[r] * dinv[c];
    const float4 v = *(const float4*)(hin + (size_t)r * OUT_DIM + q);
    float* dst = hout + (size_t)c * OUT_DIM + q;
    atomicAdd(dst + 0, w * v.x);
    atomicAdd(dst + 1, w * v.y);
    atomicAdd(dst + 2, w * v.z);
    atomicAdd(dst + 3, w * v.w);
}

// ---------------- log_softmax over 64 cols: one wave per node ----------------
__global__ __launch_bounds__(256) void k_logsoftmax(const float* __restrict__ hin,
                                                    float* __restrict__ out, int n) {
    int wave = (blockIdx.x * blockDim.x + threadIdx.x) >> 5;
    if (wave >= n) return;
    int lane = threadIdx.x & 31;
    const float* p = hin + (size_t)wave * OUT_DIM;
    float v0 = p[lane], v1 = p[lane + 32];
    float mx = fmaxf(v0, v1);
    #pragma unroll
    for (int off = 16; off > 0; off >>= 1) mx = fmaxf(mx, __shfl_xor(mx, off, 32));
    float s = expf(v0 - mx) + expf(v1 - mx);
    #pragma unroll
    for (int off = 16; off > 0; off >>= 1) s += __shfl_xor(s, off, 32);
    float lse = logf(s);
    float* o = out + (size_t)wave * OUT_DIM;
    o[lane]      = v0 - mx - lse;
    o[lane + 32] = v1 - mx - lse;
}

// ---------------- launch ----------------
extern "C" void kernel_launch(void* const* d_in, const int* in_sizes, int n_in,
                              void* d_out, int out_size, void* d_ws, size_t ws_size,
                              hipStream_t stream) {
    const float* x  = (const float*)d_in[0];
    const int*   ei = (const int*)d_in[1];
    const float* W1 = (const float*)d_in[2];
    const float* b1 = (const float*)d_in[3];
    const float* W2 = (const float*)d_in[4];
    const float* b2 = (const float*)d_in[5];

    const int N = in_sizes[0] / IN_DIM;
    const int E = in_sizes[1] / 2;
    const int* row = ei;
    const int* col = ei + E;

    auto al = [](size_t s) { return (s + 255) & ~(size_t)255; };
    char* base = (char*)d_ws;
    float*  dinv = (float*)base;   base += al((size_t)N * 4);
    __bf16* Wt1  = (__bf16*)base;  base += al((size_t)IN_DIM * HID_DIM * 2);
    __bf16* Wt2  = (__bf16*)base;  base += al((size_t)HID_DIM * OUT_DIM * 2);
    __bf16* hid  = (__bf16*)base;  base += al((size_t)N * HID_DIM * 2);
    float*  h0   = (float*)base;   base += al((size_t)N * OUT_DIM * 4);
    float*  hA   = (float*)base;   base += al((size_t)N * OUT_DIM * 4);

    // weights -> bf16 transposed
    k_prep_w<<<(IN_DIM * HID_DIM + 255) / 256, 256, 0, stream>>>(W1, Wt1, IN_DIM, HID_DIM);
    k_prep_w<<<(HID_DIM * OUT_DIM + 255) / 256, 256, 0, stream>>>(W2, Wt2, HID_DIM, OUT_DIM);

    // degrees -> dinv
    k_deg_init<<<(N + 255) / 256, 256, 0, stream>>>((unsigned*)dinv, N);
    k_deg_count<<<(E + 255) / 256, 256, 0, stream>>>((unsigned*)dinv, col, E);
    k_dinv<<<(N + 255) / 256, 256, 0, stream>>>((void*)dinv, N);

    // MLP via WMMA
    const int ntiles = N / 16;                        // 6250 for N=100000
    k_gemm1<<<(ntiles + 9) / 10, 320, 0, stream>>>(x, Wt1, b1, hid, ntiles);
    k_gemm2<<<(ntiles * 32 + 255) / 256, 256, 0, stream>>>(hid, Wt2, b2, h0, ntiles);

    // K = 5 propagation rounds, ping-pong between hA and d_out
    const float* hin = h0;
    float* bufs[2] = { hA, (float*)d_out };
    for (int it = 0; it < 5; ++it) {
        float* hout = bufs[it & 1];
        k_prop_init<<<(N * OUT_DIM + 255) / 256, 256, 0, stream>>>(hout, hin, h0, dinv, N * OUT_DIM);
        int sthreads = E * 16;
        k_prop_scatter<<<(sthreads + 255) / 256, 256, 0, stream>>>(hout, hin, row, col, dinv, E);
        hin = hout;
    }

    // final log_softmax into d_out (hin == hA after 5 rounds)
    k_logsoftmax<<<(N * 32 + 255) / 256, 256, 0, stream>>>(hin, (float*)d_out, N);
}
